// MultiHeadAttention_884763262989
// MI455X (gfx1250) — matrix operand order repaired and checked by reading, not compiled, not run on hardware
//
#include <hip/hip_runtime.h>
#include <hip/hip_bf16.h>
#include <math.h>

// ---------------------------------------------------------------------------
// MI455X (gfx1250) fused causal MHA + AdaRMSNorm/FiLM.
// Compute-bound on GEMMs -> all matmuls via v_wmma_f32_16x16x32_bf16 (f32 acc).
// Flash-style attention: scores never hit memory (saves ~0.5 GB traffic).
// GEMMs: 64x64 output per wave (4x4 WMMA tiles) -> 1 wmma per b128 load.
// Attention: 32 queries per wave -> K/V tiles shared by 2 query tiles.
// Working set (~66 MB) fits 192 MB L2.
// ---------------------------------------------------------------------------

typedef __attribute__((ext_vector_type(16))) __bf16 v16bf;
typedef __attribute__((ext_vector_type(8)))  float  v8f;
typedef __attribute__((ext_vector_type(4)))  float  f32x4;

#define BB   2
#define SS   2048
#define EE   1024
#define HH   16
#define DH   64
#define CCD  256
#define MT   (BB*SS)     // 4096 tokens
#define TWOE (2*EE)

__device__ __forceinline__ unsigned short f2bf(float x) {
  unsigned u = __float_as_uint(x);
  u += 0x7FFFu + ((u >> 16) & 1u);            // round-to-nearest-even
  return (unsigned short)(u >> 16);
}
__device__ __forceinline__ unsigned pack2(float lo, float hi) {
  return (unsigned)f2bf(lo) | ((unsigned)f2bf(hi) << 16);
}
__device__ __forceinline__ v8f vzero8() {
  v8f v;
#pragma unroll
  for (int i = 0; i < 8; ++i) v[i] = 0.0f;
  return v;
}

union ABu { f32x4 f[2]; v16bf v; };

// A-operand 16x32 bf16: half0 K=0..15, half1 K=16..31 (contiguous 32B)
__device__ __forceinline__ v16bf ldA(const unsigned short* row, int kc, int hf) {
  ABu u;
  u.f[0] = *(const f32x4*)(row + kc + hf * 16);
  u.f[1] = *(const f32x4*)(row + kc + hf * 16 + 8);
  return u.v;
}
// B-operand 32x16 bf16: half0 K=0..15, half1 K=16..31 (contiguous 32B)
__device__ __forceinline__ v16bf ldB(const unsigned short* row, int kc, int hf) {
  ABu u;
  u.f[0] = *(const f32x4*)(row + kc + hf * 16);
  u.f[1] = *(const f32x4*)(row + kc + hf * 16 + 8);
  return u.v;
}
__device__ __forceinline__ v8f wmma_bf16(v16bf a, v16bf b, v8f c) {
  return __builtin_amdgcn_wmma_f32_16x16x32_bf16(false, a, false, b,
                                                 (short)0, c, false, false);
}

// --------------------------- f32 -> bf16 convert ---------------------------
__global__ __launch_bounds__(256)
void cvt_bf16_kernel(const float* __restrict__ src, unsigned short* __restrict__ dst, int n4) {
  int i = blockIdx.x * 256 + threadIdx.x;
  if (i >= n4) return;
  f32x4 v = ((const f32x4*)src)[i];
  unsigned* d = (unsigned*)dst;
  d[2 * i]     = pack2(v[0], v[1]);
  d[2 * i + 1] = pack2(v[2], v[3]);
}

// -------------------- mod = silu(cond) @ Wc^T + bc  (B x 2E) ---------------
__global__ __launch_bounds__(256)
void mod_kernel(const float* __restrict__ cond, const float* __restrict__ Wc,
                const float* __restrict__ bc, float* __restrict__ mod) {
  __shared__ float sc[CCD];
  const int b = blockIdx.y;
  const int j = blockIdx.x * 256 + threadIdx.x;     // [0, 2E)
  float c = cond[b * CCD + threadIdx.x];
  sc[threadIdx.x] = c / (1.0f + __expf(-c));        // silu
  __syncthreads();
  float acc = bc[j];
  const float* w = Wc + (size_t)j * CCD;
#pragma unroll 4
  for (int i = 0; i < CCD; ++i) acc += sc[i] * w[i];
  mod[b * TWOE + j] = acc;
}

// ------------------ QKV projection: z=0 Q, z=1 K, z=2 V --------------------
// Q/K: D = W * X^T  -> row-major Q/K[tok][E] with contiguous bf16 stores.
// V  : D = X * Wv^T -> transposed store VT[b][e][s] (A-operand layout for PV).
// 64x64 output per wave: 16 wmma per 32-k step, 16 b128 loads.
__global__ __launch_bounds__(128)
void qkv_proj_kernel(const unsigned short* __restrict__ Xb,
                     const unsigned short* __restrict__ Wqb,
                     const unsigned short* __restrict__ Wkb,
                     const unsigned short* __restrict__ Wvb,
                     unsigned short* __restrict__ Qb,
                     unsigned short* __restrict__ Kb,
                     unsigned short* __restrict__ VTb) {
  const int lane = threadIdx.x & 31;
  const int wav  = threadIdx.x >> 5;
  const int z    = blockIdx.z;
  const int reg  = blockIdx.x * 4 + wav;            // 1024 regions of 64x64
  const int hf   = lane >> 4;
  const int lr   = lane & 15;

  const unsigned short *L, *R;
  int leftRegions;
  if (z == 0)      { L = Wqb; R = Xb;  leftRegions = EE / 64; }
  else if (z == 1) { L = Wkb; R = Xb;  leftRegions = EE / 64; }
  else             { L = Xb;  R = Wvb; leftRegions = MT / 64; }

  const int il = reg % leftRegions;
  const int ir = reg / leftRegions;
  const int l0 = il * 64;
  const int r0 = ir * 64;

  const unsigned short* Lr[4];
  const unsigned short* Rr[4];
#pragma unroll
  for (int t = 0; t < 4; ++t) {
    Lr[t] = L + (size_t)(l0 + t * 16 + lr) * EE;
    Rr[t] = R + (size_t)(r0 + t * 16 + lr) * EE;
  }

  v8f acc[4][4];
#pragma unroll
  for (int i = 0; i < 4; ++i)
#pragma unroll
    for (int j = 0; j < 4; ++j) acc[i][j] = vzero8();

  for (int kc = 0; kc < EE; kc += 32) {
    v16bf bt[4];
#pragma unroll
    for (int rt = 0; rt < 4; ++rt) bt[rt] = ldB(Rr[rt], kc, hf);
#pragma unroll
    for (int lt = 0; lt < 4; ++lt) {
      v16bf a = ldA(Lr[lt], kc, hf);
#pragma unroll
      for (int rt = 0; rt < 4; ++rt)
        acc[lt][rt] = wmma_bf16(a, bt[rt], acc[lt][rt]);
    }
  }

  // D tile element (lrow = l0+lt*16+v+hf*8, rcol = r0+rt*16+lr)
#pragma unroll
  for (int lt = 0; lt < 4; ++lt)
#pragma unroll
    for (int rt = 0; rt < 4; ++rt) {
      const v8f c = acc[lt][rt];
      uint4 o;
      o.x = pack2(c[0], c[1]); o.y = pack2(c[2], c[3]);
      o.z = pack2(c[4], c[5]); o.w = pack2(c[6], c[7]);
      const int col  = r0 + rt * 16 + lr;
      const int rowb = l0 + lt * 16 + hf * 8;
      if (z < 2) {              // col = token, rowb = feature
        unsigned short* dst = (z == 0 ? Qb : Kb);
        *(uint4*)(dst + (size_t)col * EE + rowb) = o;
      } else {                  // col = feature, rowb = token (global)
        const int b  = (rowb >= SS) ? 1 : 0;
        const int sl = rowb - b * SS;
        *(uint4*)(VTb + (size_t)b * EE * SS + (size_t)col * SS + sl) = o;
      }
    }
}

// ----------------------- flash attention (causal) --------------------------
// Per wave: one (b, h, 32-query block = 2 query tiles). S^T = K*Q^T, online
// softmax per lane (lane = query column), out^T = V^T * P^T.  K/V tile loads
// shared by both query tiles: 16 wmma per 16 b128 loads per 32-key step.
__global__ __launch_bounds__(128)
void attn_kernel(const unsigned short* __restrict__ Qb,
                 const unsigned short* __restrict__ Kb,
                 const unsigned short* __restrict__ VTb,
                 unsigned short* __restrict__ Ob) {
  const int lane = threadIdx.x & 31;
  const int wav  = threadIdx.x >> 5;
  const int qblk = blockIdx.x * 4 + wav;            // [0, S/32)
  const int h    = blockIdx.y;
  const int b    = blockIdx.z;
  const int hf   = lane >> 4;
  const int lr   = lane & 15;
  const int q0   = qblk * 32;
  const float scale = 0.125f;                       // 1/sqrt(64)

  const unsigned short* Kbase = Kb + (size_t)(b * SS) * EE + h * DH;
  const unsigned short* Vbase = VTb + (size_t)b * EE * SS + (size_t)(h * DH) * SS;

  v16bf qop[2][2];
  int sglob[2];
#pragma unroll
  for (int qt = 0; qt < 2; ++qt) {
    const unsigned short* Qrow = Qb + (size_t)(b * SS + q0 + qt * 16 + lr) * EE + h * DH;
    qop[qt][0] = ldB(Qrow, 0,  hf);                 // d = 0..31
    qop[qt][1] = ldB(Qrow, 32, hf);                 // d = 32..63
    sglob[qt]  = q0 + qt * 16 + lr;
  }

  v8f o[2][4];
#pragma unroll
  for (int qt = 0; qt < 2; ++qt)
#pragma unroll
    for (int dt = 0; dt < 4; ++dt) o[qt][dt] = vzero8();
  float m_run[2] = { -INFINITY, -INFINITY };
  float l_run[2] = { 0.0f, 0.0f };

  union PB { unsigned u[8]; v16bf v; };
  const int npair = qblk + 1;                       // 32-key steps (causal)

  for (int p = 0; p < npair; ++p) {
    const int t0 = p * 32;
    if (p + 1 < npair) {                            // global_prefetch_b8 next K pair
      __builtin_prefetch(Kbase + (size_t)(t0 + 32 + lr) * EE, 0, 3);
      __builtin_prefetch(Kbase + (size_t)(t0 + 48 + lr) * EE, 0, 3);
    }
    const unsigned short* Kr0 = Kbase + (size_t)(t0 + lr)      * EE;
    const unsigned short* Kr1 = Kbase + (size_t)(t0 + 16 + lr) * EE;
    v16bf ka[2][2];
    ka[0][0] = ldA(Kr0, 0, hf); ka[0][1] = ldA(Kr0, 32, hf);
    ka[1][0] = ldA(Kr1, 0, hf); ka[1][1] = ldA(Kr1, 32, hf);

    PB pb[2];
#pragma unroll
    for (int qt = 0; qt < 2; ++qt) {
      v8f s0 = vzero8(), s1 = vzero8();
      s0 = wmma_bf16(ka[0][0], qop[qt][0], s0);
      s0 = wmma_bf16(ka[0][1], qop[qt][1], s0);
      s1 = wmma_bf16(ka[1][0], qop[qt][0], s1);
      s1 = wmma_bf16(ka[1][1], qop[qt][1], s1);

      // scale + causal mask (lane = query sglob; rows t = t0 + v + hf*8 (+16))
      float sc0[8], sc1[8];
#pragma unroll
      for (int v = 0; v < 8; ++v) {
        const int ta = t0 + v + hf * 8;
        sc0[v] = (ta      <= sglob[qt]) ? s0[v] * scale : -INFINITY;
        sc1[v] = (ta + 16 <= sglob[qt]) ? s1[v] * scale : -INFINITY;
      }
      // per-query running max (column split across lane pair l, l^16)
      float mx = sc0[0];
#pragma unroll
      for (int v = 1; v < 8; ++v) mx = fmaxf(mx, sc0[v]);
#pragma unroll
      for (int v = 0; v < 8; ++v) mx = fmaxf(mx, sc1[v]);
      mx = fmaxf(mx, __shfl_xor(mx, 16, 32));
      const float m_new = fmaxf(m_run[qt], mx);
      const float corr  = __expf(m_run[qt] - m_new);

      float pr0[8], pr1[8], lsum = 0.0f;
#pragma unroll
      for (int v = 0; v < 8; ++v) {
        pr0[v] = __expf(sc0[v] - m_new);
        pr1[v] = __expf(sc1[v] - m_new);
        lsum += pr0[v] + pr1[v];
      }
      lsum += __shfl_xor(lsum, 16, 32);
      l_run[qt] = l_run[qt] * corr + lsum;
      m_run[qt] = m_new;

#pragma unroll
      for (int dt = 0; dt < 4; ++dt)
#pragma unroll
        for (int v = 0; v < 8; ++v) o[qt][dt][v] *= corr;

      // repack P^T (C-layout) into B-operand: pair-exchange rows via shfl_xor(16)
      float lo[8], hi[8];
#pragma unroll
      for (int r = 0; r < 8; ++r) {
        const float x0 = pr0[r], x1 = pr1[r];
        const float y0 = __shfl_xor(x0, 16, 32);
        const float y1 = __shfl_xor(x1, 16, 32);
        lo[r] = hf ? y1 : x0;   // half0: t=0..7  | half1: t=16..23
        hi[r] = hf ? x1 : y0;   // half0: t=8..15 | half1: t=24..31
      }
#pragma unroll
      for (int j = 0; j < 4; ++j) {
        pb[qt].u[j]     = pack2(lo[2 * j], lo[2 * j + 1]);
        pb[qt].u[4 + j] = pack2(hi[2 * j], hi[2 * j + 1]);
      }
    }

    // out^T += V^T * P^T  (4 d-tiles of 16; V^T tile shared by both q-tiles)
#pragma unroll
    for (int dt = 0; dt < 4; ++dt) {
      const unsigned short* Vr = Vbase + (size_t)(dt * 16 + lr) * SS + t0;
      const v16bf va = ldA(Vr, 0, hf);
#pragma unroll
      for (int qt = 0; qt < 2; ++qt)
        o[qt][dt] = wmma_bf16(va, pb[qt].v, o[qt][dt]);
    }
  }

  // finalize: divide by softmax denom, store O bf16 [b][s][h*64 + d]
#pragma unroll
  for (int qt = 0; qt < 2; ++qt) {
    const float inv = 1.0f / l_run[qt];
    unsigned short* Orow = Ob + (size_t)(b * SS + q0 + qt * 16 + lr) * EE + h * DH;
#pragma unroll
    for (int dt = 0; dt < 4; ++dt) {
      uint4 w;
      w.x = pack2(o[qt][dt][0] * inv, o[qt][dt][1] * inv);
      w.y = pack2(o[qt][dt][2] * inv, o[qt][dt][3] * inv);
      w.z = pack2(o[qt][dt][4] * inv, o[qt][dt][5] * inv);
      w.w = pack2(o[qt][dt][6] * inv, o[qt][dt][7] * inv);
      *(uint4*)(Orow + dt * 16 + hf * 8) = w;
    }
  }
}

// -------------------- out_proj = O @ Wo^T (f32 output) ---------------------
// 64x64 per wave, transposed orientation (D = Wo * O^T) for contiguous stores.
__global__ __launch_bounds__(128)
void oproj_kernel(const unsigned short* __restrict__ Ob,
                  const unsigned short* __restrict__ Wob,
                  float* __restrict__ outp) {
  const int lane = threadIdx.x & 31;
  const int wav  = threadIdx.x >> 5;
  const int reg  = blockIdx.x * 4 + wav;            // 1024 regions
  const int hf   = lane >> 4;
  const int lr   = lane & 15;
  const int il   = reg % (EE / 64);                 // Wo feature region
  const int ir   = reg / (EE / 64);                 // token region
  const int l0   = il * 64;
  const int r0   = ir * 64;

  const unsigned short* Lr[4];
  const unsigned short* Rr[4];
#pragma unroll
  for (int t = 0; t < 4; ++t) {
    Lr[t] = Wob + (size_t)(l0 + t * 16 + lr) * EE;
    Rr[t] = Ob  + (size_t)(r0 + t * 16 + lr) * EE;
  }

  v8f acc[4][4];
#pragma unroll
  for (int i = 0; i < 4; ++i)
#pragma unroll
    for (int j = 0; j < 4; ++j) acc[i][j] = vzero8();

  for (int kc = 0; kc < EE; kc += 32) {
    v16bf bt[4];
#pragma unroll
    for (int rt = 0; rt < 4; ++rt) bt[rt] = ldB(Rr[rt], kc, hf);
#pragma unroll
    for (int lt = 0; lt < 4; ++lt) {
      v16bf a = ldA(Lr[lt], kc, hf);
#pragma unroll
      for (int rt = 0; rt < 4; ++rt)
        acc[lt][rt] = wmma_bf16(a, bt[rt], acc[lt][rt]);
    }
  }

#pragma unroll
  for (int lt = 0; lt < 4; ++lt)
#pragma unroll
    for (int rt = 0; rt < 4; ++rt) {
      const v8f c = acc[lt][rt];
      const int m  = r0 + rt * 16 + lr;             // token
      const int nb = l0 + lt * 16 + hf * 8;         // feature base (8 contiguous)
      f32x4 w0, w1;
      w0[0] = c[0]; w0[1] = c[1]; w0[2] = c[2]; w0[3] = c[3];
      w1[0] = c[4]; w1[1] = c[5]; w1[2] = c[6]; w1[3] = c[7];
      *(f32x4*)(outp + (size_t)m * EE + nb)     = w0;
      *(f32x4*)(outp + (size_t)m * EE + nb + 4) = w1;
    }
}

// --------------------- AdaptiveRMSNorm + FiLM ------------------------------
__global__ __launch_bounds__(256)
void norm_film_kernel(const float* __restrict__ outp,
                      const float* __restrict__ rms_scale,
                      const float* __restrict__ mod,
                      float* __restrict__ y) {
  __shared__ float red[256];
  const int row = blockIdx.x;                       // [0, MT)
  const int b   = row / SS;
  const int t   = threadIdx.x;
  const float* xr = outp + (size_t)row * EE;
  f32x4 v = *(const f32x4*)(xr + t * 4);
  red[t] = v[0] * v[0] + v[1] * v[1] + v[2] * v[2] + v[3] * v[3];
  __syncthreads();
  for (int s = 128; s > 0; s >>= 1) {
    if (t < s) red[t] += red[t + s];
    __syncthreads();
  }
  const float inv = rsqrtf(red[0] * (1.0f / EE) + 1e-6f);
  const float* modb = mod + b * TWOE;
  f32x4 o;
#pragma unroll
  for (int i = 0; i < 4; ++i) {
    const int e = t * 4 + i;
    o[i] = v[i] * inv * rms_scale[e] * (1.0f + modb[EE + e]) + modb[e];
  }
  *(f32x4*)(y + (size_t)row * EE + t * 4) = o;
}

// ---------------------------------------------------------------------------
extern "C" void kernel_launch(void* const* d_in, const int* in_sizes, int n_in,
                              void* d_out, int out_size, void* d_ws, size_t ws_size,
                              hipStream_t stream) {
  (void)in_sizes; (void)n_in; (void)out_size; (void)ws_size;
  const float* x     = (const float*)d_in[0];
  /* d_in[1] = causal mask: structure known, not needed */
  const float* cond  = (const float*)d_in[2];
  const float* Wq    = (const float*)d_in[3];
  const float* Wk    = (const float*)d_in[4];
  const float* Wv    = (const float*)d_in[5];
  const float* Wo    = (const float*)d_in[6];
  const float* rmssc = (const float*)d_in[7];
  const float* Wc    = (const float*)d_in[8];
  const float* bc    = (const float*)d_in[9];
  float* y = (float*)d_out;

  char* p = (char*)d_ws;
  unsigned short* Xb  = (unsigned short*)p; p += (size_t)MT * EE * 2;
  unsigned short* Wqb = (unsigned short*)p; p += (size_t)EE * EE * 2;
  unsigned short* Wkb = (unsigned short*)p; p += (size_t)EE * EE * 2;
  unsigned short* Wvb = (unsigned short*)p; p += (size_t)EE * EE * 2;
  unsigned short* Wob = (unsigned short*)p; p += (size_t)EE * EE * 2;
  unsigned short* Qb  = (unsigned short*)p; p += (size_t)MT * EE * 2;
  unsigned short* Kb  = (unsigned short*)p; p += (size_t)MT * EE * 2;
  unsigned short* VTb = (unsigned short*)p; p += (size_t)MT * EE * 2;
  unsigned short* Ob  = (unsigned short*)p; p += (size_t)MT * EE * 2;
  float*          op  = (float*)p;          p += (size_t)MT * EE * 4;
  float*          mw  = (float*)p;          p += (size_t)BB * TWOE * 4;

  // bf16 staging
  cvt_bf16_kernel<<<(MT * EE / 4 + 255) / 256, 256, 0, stream>>>(x,  Xb,  MT * EE / 4);
  cvt_bf16_kernel<<<(EE * EE / 4 + 255) / 256, 256, 0, stream>>>(Wq, Wqb, EE * EE / 4);
  cvt_bf16_kernel<<<(EE * EE / 4 + 255) / 256, 256, 0, stream>>>(Wk, Wkb, EE * EE / 4);
  cvt_bf16_kernel<<<(EE * EE / 4 + 255) / 256, 256, 0, stream>>>(Wv, Wvb, EE * EE / 4);
  cvt_bf16_kernel<<<(EE * EE / 4 + 255) / 256, 256, 0, stream>>>(Wo, Wob, EE * EE / 4);

  // FiLM modulation (tiny)
  mod_kernel<<<dim3(TWOE / 256, BB), 256, 0, stream>>>(cond, Wc, bc, mw);

  // QKV projections (WMMA, 64x64 per wave)
  qkv_proj_kernel<<<dim3(256, 1, 3), 128, 0, stream>>>(Xb, Wqb, Wkb, Wvb, Qb, Kb, VTb);

  // flash attention (WMMA, 32 queries per wave)
  attn_kernel<<<dim3(SS / 32 / 4, HH, BB), 128, 0, stream>>>(Qb, Kb, VTb, Ob);

  // output projection (WMMA, 64x64 per wave)
  oproj_kernel<<<dim3(256), 128, 0, stream>>>(Ob, Wob, op);

  // RMSNorm + FiLM
  norm_film_kernel<<<dim3(MT), 256, 0, stream>>>(op, rmssc, mw, y);
}
